// HybridModel_71863392797059
// MI455X (gfx1250) — matrix operand order repaired and checked by reading, not compiled, not run on hardware
//
#include <hip/hip_runtime.h>
#include <hip/hip_bf16.h>

typedef __attribute__((ext_vector_type(16))) _Float16 v16h;
typedef __attribute__((ext_vector_type(8)))  float    v8f;

#define NQ     12
#define NSTATE 4096
#define BATCH  512
#define DIN    768
#define H1     256
#define NLAB   28

// ---------------------------------------------------------------------------
// Kernel A: h = tanh(x @ proj_w^T + proj_b), then LayerNorm(12) -> hn
// One wave per 16-row strip. WMMA f32<-f16 16x16x32, K=768 (24 k-tiles).
// ---------------------------------------------------------------------------
__global__ __launch_bounds__(32) void proj_tanh_ln_kernel(
    const float* __restrict__ x, const float* __restrict__ pw,
    const float* __restrict__ pb, const float* __restrict__ lnw,
    const float* __restrict__ lnb, float* __restrict__ hn)
{
    __shared__ float hbuf[16 * 16];
    const int lane  = threadIdx.x;
    const int mtile = blockIdx.x;          // 16 rows of the 512-row batch
    const int row   = lane & 15;           // A-row / B-col / D-col index
    const int half  = lane >> 4;
    const int browc = (row < NQ) ? row : 0;  // clamped B row (proj_w row)

    v8f c = {};
    for (int kt = 0; kt < DIN / 32; ++kt) {
        v16h a, b;
        const int ka = kt * 32 + half * 16;
        const float* xr = x + (size_t)(mtile * 16 + row) * DIN + ka;
        #pragma unroll
        for (int j = 0; j < 8; ++j) {
            a[j]     = (_Float16)xr[j];
            a[j + 8] = (_Float16)xr[8 + j];
        }
        const int kb = kt * 32 + half * 16;
        const float* wr = pw + (size_t)browc * DIN + kb;
        #pragma unroll
        for (int j = 0; j < 16; ++j) b[j] = (_Float16)wr[j];
        c = __builtin_amdgcn_wmma_f32_16x16x32_f16(false, a, false, b,
                                                   (short)0, c, false, false);
    }
    const float bias = pb[browc];
    #pragma unroll
    for (int r = 0; r < 8; ++r) {
        int m = (lane < 16) ? r : r + 8;
        hbuf[m * 16 + row] = tanhf(c[r] + bias);
    }
    __syncthreads();
    if (lane < 16) {
        float mu = 0.f;
        #pragma unroll
        for (int j = 0; j < NQ; ++j) mu += hbuf[lane * 16 + j];
        mu *= (1.f / NQ);
        float var = 0.f;
        #pragma unroll
        for (int j = 0; j < NQ; ++j) {
            float d = hbuf[lane * 16 + j] - mu; var += d * d;
        }
        var *= (1.f / NQ);
        float rs = rsqrtf(var + 1e-5f);
        const int grow = mtile * 16 + lane;
        #pragma unroll
        for (int j = 0; j < NQ; ++j)
            hn[grow * NQ + j] = (hbuf[lane * 16 + j] - mu) * rs * lnw[j] + lnb[j];
    }
}

// ---------------------------------------------------------------------------
// Kernel B: 12-qubit statevector sim, one block (256 thr = 8 wave32) per
// sample; 32KB state resident in LDS.
//
// Pass-count reductions (all exact):
//  - CNOT ring folded into GF(2) index maps (col[] / rinv[]) -> 0 passes.
//  - RY(w) fused with layer-0 Rot(w) (they commute across wires): 24 -> 12.
//  - Measurement fused into the final gate pass: last pass reads pairs,
//    accumulates <Z_w> via rinv_final parity masks, and skips the writeback.
//  Total: 84 passes (naive) -> 35.5 effective full-state LDS passes.
// ---------------------------------------------------------------------------
__global__ __launch_bounds__(256) void qsim_kernel(
    const float* __restrict__ hn, const float* __restrict__ qw,
    float* __restrict__ qout)
{
    __shared__ float2 st[NSTATE];        // 32 KB statevector
    __shared__ float  gm[48 * 8];        // gate matrices (complex 2x2)
    __shared__ float  zacc[NQ];
    const int tid = threadIdx.x;
    const int s   = blockIdx.x;

    // |0...0> init
    #pragma unroll
    for (int j = 0; j < 16; ++j) st[tid + 256 * j] = make_float2(0.f, 0.f);
    if (tid == 0) st[0] = make_float2(1.f, 0.f);
    if (tid < NQ) zacc[tid] = 0.f;

    // gm[0..11] = RY(hn[w]);  gm[12+12l+w] = Rot(q_w[l,w,:])
    if (tid < 12) {
        float th = hn[s * NQ + tid];
        float ch = cosf(0.5f * th), sh = sinf(0.5f * th);
        float* m = gm + tid * 8;
        m[0] = ch; m[1] = 0.f; m[2] = -sh; m[3] = 0.f;
        m[4] = sh; m[5] = 0.f; m[6] =  ch; m[7] = 0.f;
    } else if (tid < 48) {
        int l = (tid - 12) / 12, w = (tid - 12) % 12;
        float phi = qw[(l * NQ + w) * 3 + 0];
        float th  = qw[(l * NQ + w) * 3 + 1];
        float om  = qw[(l * NQ + w) * 3 + 2];
        float ct = cosf(0.5f * th), sn = sinf(0.5f * th);
        float a1 = 0.5f * (phi + om), a2 = 0.5f * (phi - om);
        float c1 = cosf(a1), s1 = sinf(a1), c2 = cosf(a2), s2 = sinf(a2);
        float* m = gm + tid * 8;
        m[0] =  c1 * ct; m[1] = -s1 * ct;   // ep*c
        m[2] = -c2 * sn; m[3] = -s2 * sn;   // -conj(em)*s
        m[4] =  c2 * sn; m[5] = -s2 * sn;   // em*s
        m[6] =  c1 * ct; m[7] =  s1 * ct;   // conj(ep)*c
    }
    __syncthreads();
    // Fuse: gm[12+w] := Rot(0,w) * RY(w)   (complex 2x2 product, per wire)
    if (tid < 12) {
        const float* A = gm + tid * 8;          // RY (applied first)
        float*       B = gm + (12 + tid) * 8;   // Rot layer 0
        float F[8];
        F[0] = B[0]*A[0] - B[1]*A[1] + B[2]*A[4] - B[3]*A[5];
        F[1] = B[0]*A[1] + B[1]*A[0] + B[2]*A[5] + B[3]*A[4];
        F[2] = B[0]*A[2] - B[1]*A[3] + B[2]*A[6] - B[3]*A[7];
        F[3] = B[0]*A[3] + B[1]*A[2] + B[2]*A[7] + B[3]*A[6];
        F[4] = B[4]*A[0] - B[5]*A[1] + B[6]*A[4] - B[7]*A[5];
        F[5] = B[4]*A[1] + B[5]*A[0] + B[6]*A[5] + B[7]*A[4];
        F[6] = B[4]*A[2] - B[5]*A[3] + B[6]*A[6] - B[7]*A[7];
        F[7] = B[4]*A[3] + B[5]*A[2] + B[6]*A[7] + B[7]*A[6];
        #pragma unroll
        for (int j = 0; j < 8; ++j) B[j] = F[j];
    }
    __syncthreads();

    // 1q gate on logical bit: physical pair offset `mask`, role from parity
    // of p & rinv. 2048 pairs / 256 threads, barrier per pass.
    auto apply1q = [&](int g, unsigned mask, unsigned rinv) {
        const float m00r = gm[g*8+0], m00i = gm[g*8+1];
        const float m01r = gm[g*8+2], m01i = gm[g*8+3];
        const float m10r = gm[g*8+4], m10i = gm[g*8+5];
        const float m11r = gm[g*8+6], m11i = gm[g*8+7];
        const int pb = __builtin_ctz(mask);
        #pragma unroll
        for (int k = 0; k < 8; ++k) {
            unsigned p  = (unsigned)tid + 256u * k;
            unsigned pa = ((p >> pb) << (pb + 1)) | (p & ((1u << pb) - 1u));
            unsigned pc = pa ^ mask;
            unsigned par = __popc(pa & rinv) & 1u;
            unsigned lo = par ? pc : pa;
            unsigned hi = par ? pa : pc;
            float2 s0 = st[lo], s1v = st[hi], n0, n1;
            n0.x = m00r*s0.x - m00i*s0.y + m01r*s1v.x - m01i*s1v.y;
            n0.y = m00r*s0.y + m00i*s0.x + m01r*s1v.y + m01i*s1v.x;
            n1.x = m10r*s0.x - m10i*s0.y + m11r*s1v.x - m11i*s1v.y;
            n1.y = m10r*s0.y + m10i*s0.x + m11r*s1v.y + m11i*s1v.x;
            st[lo] = n0; st[hi] = n1;
        }
        __syncthreads();
    };

    // GF(2) logical->physical map state (uniform; data-independent)
    unsigned col[NQ], rinv[NQ];
    #pragma unroll
    for (int b = 0; b < NQ; ++b) { col[b] = 1u << b; rinv[b] = 1u << b; }

    // Fused RY+Rot layer 0 (identity map)
    for (int w = 0; w < NQ; ++w) apply1q(12 + w, col[w], rinv[w]);
    // fold CNOT ring r=1
    #pragma unroll
    for (int w = 0; w < NQ; ++w) {
        const int t = (w + 1) % NQ; col[w] ^= col[t]; rinv[t] ^= rinv[w];
    }
    // Rot layer 1
    for (int w = 0; w < NQ; ++w) apply1q(24 + w, col[w], rinv[w]);
    // fold CNOT ring r=2
    #pragma unroll
    for (int w = 0; w < NQ; ++w) {
        const int t = (w + 2) % NQ; col[w] ^= col[t]; rinv[t] ^= rinv[w];
    }
    // Rot layer 2, gates 0..10 (gate 11 fused with measurement below)
    for (int w = 0; w < NQ - 1; ++w) apply1q(36 + w, col[w], rinv[w]);

    // rinv after folding the final CNOT ring (r=3): measurement-basis masks
    unsigned rf[NQ];
    #pragma unroll
    for (int b = 0; b < NQ; ++b) rf[b] = rinv[b];
    #pragma unroll
    for (int w = 0; w < NQ; ++w) { const int t = (w + 3) % NQ; rf[t] ^= rf[w]; }

    // Final gate (layer 2, w=11) fused with <Z> accumulation: read-only pass,
    // no writeback (state is dead afterwards).
    float z[NQ];
    #pragma unroll
    for (int w = 0; w < NQ; ++w) z[w] = 0.f;
    {
        const int g = 47;
        const float m00r = gm[g*8+0], m00i = gm[g*8+1];
        const float m01r = gm[g*8+2], m01i = gm[g*8+3];
        const float m10r = gm[g*8+4], m10i = gm[g*8+5];
        const float m11r = gm[g*8+6], m11i = gm[g*8+7];
        const unsigned mask = col[NQ - 1], ri = rinv[NQ - 1];
        const int pb = __builtin_ctz(mask);
        #pragma unroll
        for (int k = 0; k < 8; ++k) {
            unsigned p  = (unsigned)tid + 256u * k;
            unsigned pa = ((p >> pb) << (pb + 1)) | (p & ((1u << pb) - 1u));
            unsigned pc = pa ^ mask;
            unsigned par = __popc(pa & ri) & 1u;
            unsigned lo = par ? pc : pa;
            unsigned hi = par ? pa : pc;
            float2 s0 = st[lo], s1v = st[hi], n0, n1;
            n0.x = m00r*s0.x - m00i*s0.y + m01r*s1v.x - m01i*s1v.y;
            n0.y = m00r*s0.y + m00i*s0.x + m01r*s1v.y + m01i*s1v.x;
            n1.x = m10r*s0.x - m10i*s0.y + m11r*s1v.x - m11i*s1v.y;
            n1.y = m10r*s0.y + m10i*s0.x + m11r*s1v.y + m11i*s1v.x;
            float p0 = n0.x * n0.x + n0.y * n0.y;
            float p1 = n1.x * n1.x + n1.y * n1.y;
            #pragma unroll
            for (int w = 0; w < NQ; ++w) {
                z[w] += (__popc(lo & rf[w]) & 1) ? -p0 : p0;
                z[w] += (__popc(hi & rf[w]) & 1) ? -p1 : p1;
            }
        }
    }
    #pragma unroll
    for (int w = 0; w < NQ; ++w) atomicAdd(&zacc[w], z[w]);  // ds_add_f32
    __syncthreads();
    if (tid < NQ) qout[s * NQ + tid] = zacc[tid];
}

// ---------------------------------------------------------------------------
// Kernel C: out = relu(q @ h1_w^T + b1) @ h2_w^T + b2, fused via LDS z-strip.
// ---------------------------------------------------------------------------
__global__ __launch_bounds__(32) void mlp_kernel(
    const float* __restrict__ q,  const float* __restrict__ w1,
    const float* __restrict__ b1, const float* __restrict__ w2,
    const float* __restrict__ b2, float* __restrict__ out)
{
    __shared__ _Float16 zbuf[16 * H1];     // 8 KB f16 z strip
    const int lane  = threadIdx.x;
    const int mtile = blockIdx.x;
    const int row   = lane & 15;
    const int half  = lane >> 4;

    v16h a = {};
    {
        const int kb = half * 16;
        const float* qr = q + (size_t)(mtile * 16 + row) * NQ;
        #pragma unroll
        for (int j = 0; j < 8; ++j) {
            int k0 = kb + j, k1 = kb + 8 + j;
            a[j]     = (k0 < NQ) ? (_Float16)qr[k0] : (_Float16)0.f;
            a[j + 8] = (k1 < NQ) ? (_Float16)qr[k1] : (_Float16)0.f;
        }
    }
    for (int nt = 0; nt < H1 / 16; ++nt) {
        v16h b = {};
        const int n  = nt * 16 + row;
        const int kb = half * 16;
        if (kb < NQ) {
            const float* wr = w1 + (size_t)n * NQ;
            #pragma unroll
            for (int j = 0; j < 16; ++j)
                b[j] = (kb + j < NQ) ? (_Float16)wr[kb + j] : (_Float16)0.f;
        }
        v8f c = {};
        c = __builtin_amdgcn_wmma_f32_16x16x32_f16(false, a, false, b,
                                                   (short)0, c, false, false);
        #pragma unroll
        for (int r = 0; r < 8; ++r) {
            int m = (lane < 16) ? r : r + 8;
            float v = c[r] + b1[n];
            v = v > 0.f ? v : 0.f;
            zbuf[m * H1 + n] = (_Float16)v;
        }
    }
    __syncthreads();

    #pragma unroll
    for (int nt = 0; nt < 2; ++nt) {
        v8f c = {};
        const int n   = nt * 16 + row;
        const bool nv = (n < NLAB);
        const int nc  = nv ? n : 0;
        for (int kt = 0; kt < H1 / 32; ++kt) {
            v16h af, bf;
            const int ka = kt * 32 + half * 16;
            const _Float16* zr = zbuf + row * H1 + ka;
            #pragma unroll
            for (int j = 0; j < 8; ++j) { af[j] = zr[j]; af[j + 8] = zr[8 + j]; }
            const int kb = kt * 32 + half * 16;
            const float* wr = w2 + (size_t)nc * H1 + kb;
            #pragma unroll
            for (int j = 0; j < 16; ++j) bf[j] = (_Float16)wr[j];
            c = __builtin_amdgcn_wmma_f32_16x16x32_f16(false, af, false, bf,
                                                       (short)0, c, false, false);
        }
        if (nv) {
            float bias = b2[n];
            #pragma unroll
            for (int r = 0; r < 8; ++r) {
                int m = (lane < 16) ? r : r + 8;
                out[(size_t)(mtile * 16 + m) * NLAB + n] = c[r] + bias;
            }
        }
    }
}

// ---------------------------------------------------------------------------
extern "C" void kernel_launch(void* const* d_in, const int* in_sizes, int n_in,
                              void* d_out, int out_size, void* d_ws, size_t ws_size,
                              hipStream_t stream)
{
    const float* x   = (const float*)d_in[0];
    const float* pw  = (const float*)d_in[1];
    const float* pb  = (const float*)d_in[2];
    const float* lnw = (const float*)d_in[3];
    const float* lnb = (const float*)d_in[4];
    const float* qw  = (const float*)d_in[5];
    const float* w1  = (const float*)d_in[6];
    const float* b1  = (const float*)d_in[7];
    const float* w2  = (const float*)d_in[8];
    const float* b2  = (const float*)d_in[9];
    float* out = (float*)d_out;

    float* hn = (float*)d_ws;              // 512*12 f32
    float* q  = hn + BATCH * NQ;           // 512*12 f32

    proj_tanh_ln_kernel<<<BATCH / 16, 32, 0, stream>>>(x, pw, pb, lnw, lnb, hn);
    qsim_kernel<<<BATCH, 256, 0, stream>>>(hn, qw, q);
    mlp_kernel<<<BATCH / 16, 32, 0, stream>>>(q, w1, b1, w2, b2, out);
}
